// EnergDev_5257039970318
// MI455X (gfx1250) — matrix operand order repaired and checked by reading, not compiled, not run on hardware
//
#include <hip/hip_runtime.h>
#include <hip/hip_bf16.h>

typedef __attribute__((ext_vector_type(16))) _Float16 v16h;
typedef __attribute__((ext_vector_type(8)))  float    v8f;

__device__ __forceinline__ float lsilu_f(float x, float a) {
    return x * (1.0f / (1.0f + __expf(-x))) + a * x;
}

__device__ __forceinline__ v8f wmma_f16(v16h a, v16h b, v8f c) {
    // D = A(16x32 f16) * B(32x16 f16) + C(16x16 f32)
    return __builtin_amdgcn_wmma_f32_16x16x32_f16(false, a, false, b, (short)0, c, false, false);
}

// A-fragment K-pair base for vgpr v (0..7), lane-half h: pairs (kb, kb+1)
__device__ __forceinline__ int a_kbase(int v, int half) {
    return (v << 1) + (half << 4);
}

// ---------------------------------------------------------------------------
// Weight repack: W[K,N] f32 -> fragment-ordered f16.
// GEMM reads: b = *(v16h*)&dst[(((nt*ksteps)+ks)*32 + lane)*16]
// element i of lane -> B[k = ks*32 + i + 16*(lane>>4)][n = nt*16 + (lane&15)]
// ---------------------------------------------------------------------------
__global__ __launch_bounds__(256) void pack_kernel(const float* __restrict__ W,
                                                   _Float16* __restrict__ dst,
                                                   int K, int N, int ksteps) {
    int t = blockIdx.x * 256 + threadIdx.x;
    int total = (N / 16) * ksteps * 512;
    if (t >= total) return;
    int i    = t & 15;
    int lane = (t >> 4) & 31;
    int ks   = (t >> 9) % ksteps;
    int nt   = t / (ksteps * 512);
    int half = lane >> 4, col = lane & 15;
    int k = ks * 32 + i + 16 * half;
    int n = nt * 16 + col;
    dst[t] = (k < K) ? (_Float16)W[(size_t)k * N + n] : (_Float16)0.0f;
}

// ---------------------------------------------------------------------------
// prep: nf = [x[:,0], x[:,1:4] @ matrix[batch]]
// ---------------------------------------------------------------------------
__global__ __launch_bounds__(256) void prep_kernel(const float* __restrict__ x,
                                                   const float* __restrict__ matrix,
                                                   const int* __restrict__ batch,
                                                   float* __restrict__ out, int N) {
    int n = blockIdx.x * 256 + threadIdx.x;
    if (n >= N) return;
    const float* xr = x + (size_t)n * 4;
    const float* M  = matrix + (size_t)batch[n] * 9;
    float f0 = xr[1], f1 = xr[2], f2 = xr[3];
    float* o = out + (size_t)n * 4;
    o[0] = xr[0];
    o[1] = f0 * M[0] + f1 * M[3] + f2 * M[6];
    o[2] = f0 * M[1] + f1 * M[4] + f2 * M[7];
    o[3] = f0 * M[2] + f1 * M[5] + f2 * M[8];
}

__global__ __launch_bounds__(256) void zero_kernel(float* p, long n) {
    long i = (long)blockIdx.x * 256 + threadIdx.x;
    if (i < n) p[i] = 0.0f;
}

__global__ __launch_bounds__(256) void finalize_kernel(float* p, long n) {
    long i = (long)blockIdx.x * 256 + threadIdx.x;
    if (i < n) p[i] = lsilu_f(p[i], 0.1f);
}

// ---------------------------------------------------------------------------
// Edge conv: msg = mlp2(concat(feat[dst], feat[src])); acc[dst] += msg
// One wave handles 16 edges. DH = 64 always.
// ---------------------------------------------------------------------------
template <int DIN, int DOUT>
__global__ __launch_bounds__(256) void conv_kernel(const float* __restrict__ feat,
                                                   const int* __restrict__ esrc,
                                                   const int* __restrict__ edst,
                                                   const _Float16* __restrict__ w1p,
                                                   const float* __restrict__ b1,
                                                   const _Float16* __restrict__ w2p,
                                                   const float* __restrict__ b2,
                                                   float* __restrict__ acc, int nEdges) {
    constexpr int DH  = 64;
    constexpr int W   = DIN / 2;
    constexpr int KS1 = (DIN + 31) / 32;
    constexpr int KS2 = DH / 32;
    __shared__ _Float16 hbuf[8][16 * DH];

    const int wave = threadIdx.x >> 5;
    const int lane = threadIdx.x & 31;
    const int half = lane >> 4;
    const int ln15 = lane & 15;
    const long tile = (long)blockIdx.x * 8 + wave;
    const long e0 = tile * 16;
    if (e0 >= nEdges) return;            // wave-uniform; WMMA EXEC stays all-ones

    // this lane's A-row edge
    const int eA = (int)e0 + ln15;
    const float* rowD = feat + (size_t)edst[eA] * W;
    const float* rowS = feat + (size_t)esrc[eA] * W;

    v8f hacc[DH / 16];
#pragma unroll
    for (int t = 0; t < DH / 16; ++t)
#pragma unroll
        for (int j = 0; j < 8; ++j) hacc[t][j] = 0.0f;

    for (int ks = 0; ks < KS1; ++ks) {
        v16h a;
#pragma unroll
        for (int v = 0; v < 8; ++v) {
            int gk = ks * 32 + a_kbase(v, half);
            float x0 = 0.0f, x1 = 0.0f;
            if (gk < DIN) {
                const float* p = (gk < W) ? (rowD + gk) : (rowS + gk - W);
                x0 = p[0]; x1 = p[1];
            }
            a[2 * v]     = (_Float16)x0;
            a[2 * v + 1] = (_Float16)x1;
        }
#pragma unroll
        for (int nt = 0; nt < DH / 16; ++nt) {
            v16h b = *(const v16h*)(w1p + (((size_t)nt * KS1 + ks) * 32 + lane) * 16);
            hacc[nt] = wmma_f16(a, b, hacc[nt]);
        }
    }

    // bias + lsilu(0.05) -> LDS (f16), same-wave DS ordering needs no barrier
#pragma unroll
    for (int nt = 0; nt < DH / 16; ++nt)
#pragma unroll
        for (int r = 0; r < 8; ++r) {
            int m = r + 8 * half;
            int n = nt * 16 + ln15;
            hbuf[wave][m * DH + n] = (_Float16)lsilu_f(hacc[nt][r] + b1[n], 0.05f);
        }

    v8f oacc[DOUT / 16];
#pragma unroll
    for (int t = 0; t < DOUT / 16; ++t)
#pragma unroll
        for (int j = 0; j < 8; ++j) oacc[t][j] = 0.0f;

    for (int ks = 0; ks < KS2; ++ks) {
        v16h a;
#pragma unroll
        for (int v = 0; v < 8; ++v) {
            int k = ks * 32 + a_kbase(v, half);
            a[2 * v]     = hbuf[wave][ln15 * DH + k];
            a[2 * v + 1] = hbuf[wave][ln15 * DH + k + 1];
        }
#pragma unroll
        for (int nt = 0; nt < DOUT / 16; ++nt) {
            v16h b = *(const v16h*)(w2p + (((size_t)nt * KS2 + ks) * 32 + lane) * 16);
            oacc[nt] = wmma_f16(a, b, oacc[nt]);
        }
    }

    int dstn[8];
#pragma unroll
    for (int r = 0; r < 8; ++r) dstn[r] = edst[(int)e0 + r + 8 * half];

#pragma unroll
    for (int nt = 0; nt < DOUT / 16; ++nt)
#pragma unroll
        for (int r = 0; r < 8; ++r) {
            int n = nt * 16 + ln15;
            float val = lsilu_f(oacc[nt][r] + b2[n], 0.05f);
            atomicAdd(acc + (size_t)dstn[r] * DOUT + n, val);
        }
}

// ---------------------------------------------------------------------------
// Node MLP: out = mlp2(feat) + feat.  DH = 128, DOUT = DIN.
// ---------------------------------------------------------------------------
template <int DIN>
__global__ __launch_bounds__(256) void inter_kernel(const float* __restrict__ feat,
                                                    const _Float16* __restrict__ w1p,
                                                    const float* __restrict__ b1,
                                                    const _Float16* __restrict__ w2p,
                                                    const float* __restrict__ b2,
                                                    float* __restrict__ out, int nNodes) {
    constexpr int DH  = 128;
    constexpr int KS1 = DIN / 32;   // 1, 4, 2
    constexpr int KS2 = DH / 32;    // 4
    __shared__ _Float16 hbuf[8][16 * DH];

    const int wave = threadIdx.x >> 5;
    const int lane = threadIdx.x & 31;
    const int half = lane >> 4;
    const int ln15 = lane & 15;
    const long tile = (long)blockIdx.x * 8 + wave;
    const long n0 = tile * 16;
    if (n0 >= nNodes) return;

    const float* rowA = feat + (size_t)(n0 + ln15) * DIN;

    v8f hacc[DH / 16];
#pragma unroll
    for (int t = 0; t < DH / 16; ++t)
#pragma unroll
        for (int j = 0; j < 8; ++j) hacc[t][j] = 0.0f;

    for (int ks = 0; ks < KS1; ++ks) {
        v16h a;
#pragma unroll
        for (int v = 0; v < 8; ++v) {
            int gk = ks * 32 + a_kbase(v, half);
            a[2 * v]     = (_Float16)rowA[gk];
            a[2 * v + 1] = (_Float16)rowA[gk + 1];
        }
#pragma unroll
        for (int nt = 0; nt < DH / 16; ++nt) {
            v16h b = *(const v16h*)(w1p + (((size_t)nt * KS1 + ks) * 32 + lane) * 16);
            hacc[nt] = wmma_f16(a, b, hacc[nt]);
        }
    }

#pragma unroll
    for (int nt = 0; nt < DH / 16; ++nt)
#pragma unroll
        for (int r = 0; r < 8; ++r) {
            int m = r + 8 * half;
            int n = nt * 16 + ln15;
            hbuf[wave][m * DH + n] = (_Float16)lsilu_f(hacc[nt][r] + b1[n], 0.05f);
        }

    v8f oacc[DIN / 16];
#pragma unroll
    for (int t = 0; t < DIN / 16; ++t)
#pragma unroll
        for (int j = 0; j < 8; ++j) oacc[t][j] = 0.0f;

    for (int ks = 0; ks < KS2; ++ks) {
        v16h a;
#pragma unroll
        for (int v = 0; v < 8; ++v) {
            int k = ks * 32 + a_kbase(v, half);
            a[2 * v]     = hbuf[wave][ln15 * DH + k];
            a[2 * v + 1] = hbuf[wave][ln15 * DH + k + 1];
        }
#pragma unroll
        for (int nt = 0; nt < DIN / 16; ++nt) {
            v16h b = *(const v16h*)(w2p + (((size_t)nt * KS2 + ks) * 32 + lane) * 16);
            oacc[nt] = wmma_f16(a, b, oacc[nt]);
        }
    }

#pragma unroll
    for (int nt = 0; nt < DIN / 16; ++nt)
#pragma unroll
        for (int r = 0; r < 8; ++r) {
            int m = r + 8 * half;
            int n = nt * 16 + ln15;
            size_t idx = (size_t)(n0 + m) * DIN + n;
            out[idx] = lsilu_f(oacc[nt][r] + b2[n], 0.05f) + feat[idx];
        }
}

// ---------------------------------------------------------------------------
// segment pool + final FC
// ---------------------------------------------------------------------------
__global__ __launch_bounds__(256) void pool_kernel(const float* __restrict__ feat,
                                                   const int* __restrict__ batch,
                                                   float* __restrict__ pooled, int N) {
    long t = (long)blockIdx.x * 256 + threadIdx.x;
    if (t >= (long)N * 128) return;
    int n = (int)(t >> 7);
    int j = (int)(t & 127);
    atomicAdd(&pooled[(size_t)batch[n] * 128 + j], feat[t]);
}

__global__ __launch_bounds__(256) void fc_kernel(const float* __restrict__ pooled,
                                                 const float* __restrict__ Wf,
                                                 const float* __restrict__ bf,
                                                 float* __restrict__ out) {
    int g = threadIdx.x;
    if (g >= 256) return;
    float s = bf[0];
#pragma unroll 8
    for (int j = 0; j < 128; ++j) s += pooled[(size_t)g * 128 + j] * Wf[j];
    out[g] = -lsilu_f(s, 0.1f) * 0.1f;
}

// ---------------------------------------------------------------------------
extern "C" void kernel_launch(void* const* d_in, const int* in_sizes, int n_in,
                              void* d_out, int out_size, void* d_ws, size_t ws_size,
                              hipStream_t stream) {
    const float* x      = (const float*)d_in[0];
    const float* matrix = (const float*)d_in[1];

    // params: conv1..conv4 occupy slots 2..17 (W1,b1,W2,b2 each).
    // After that, either insertion order (inter1,inter2,inter3,fc1) or
    // JAX sorted-key order (fc1, inter1, inter2, inter3). Detect via sizes.
    int i_c1 = 2, i_c2 = 6, i_c3 = 10, i_c4 = 14;
    int i_fc, i_in1, i_in2, i_in3;
    bool fc_first = (n_in >= 20 && in_sizes[18] == 128 && in_sizes[19] == 1);
    if (fc_first) { i_fc = 18; i_in1 = 20; i_in2 = 24; i_in3 = 28; }
    else          { i_in1 = 18; i_in2 = 22; i_in3 = 26; i_fc = 30; }
    const int i_edge = 32, i_batch = 33;

    const int N = in_sizes[0] / 4;
    const int E = in_sizes[i_edge] / 2;
    const int* esrc  = (const int*)d_in[i_edge];
    const int* edst  = esrc + E;
    const int* batch = (const int*)d_in[i_batch];

    // workspace carve
    char* ws = (char*)d_ws;
    size_t off = 0;
    auto allocH = [&](size_t elems) -> _Float16* {
        _Float16* p = (_Float16*)(ws + off); off += elems * 2; return p;
    };
    _Float16* c1w1 = allocH(2048);  _Float16* c1w2 = allocH(2048);
    _Float16* c2w1 = allocH(4096);  _Float16* c2w2 = allocH(8192);
    _Float16* c3w1 = allocH(16384); _Float16* c3w2 = allocH(4096);
    _Float16* c4w1 = allocH(8192);  _Float16* c4w2 = allocH(8192);
    _Float16* i1w1 = allocH(4096);  _Float16* i1w2 = allocH(4096);
    _Float16* i2w1 = allocH(16384); _Float16* i2w2 = allocH(16384);
    _Float16* i3w1 = allocH(8192);  _Float16* i3w2 = allocH(8192);
    off = (off + 255) & ~(size_t)255;
    float* buf0   = (float*)(ws + off); off += (size_t)N * 128 * 4;
    float* buf1   = (float*)(ws + off); off += (size_t)N * 128 * 4;
    float* pooled = (float*)(ws + off); off += 256 * 128 * 4;

    auto bias = [&](int idx) { return (const float*)d_in[idx]; };
    auto packW = [&](int idx, _Float16* dst, int K, int Nn) {
        int ksteps = (K + 31) / 32;
        int total = (Nn / 16) * ksteps * 512;
        pack_kernel<<<(total + 255) / 256, 256, 0, stream>>>((const float*)d_in[idx], dst, K, Nn, ksteps);
    };
    packW(i_c1 + 0, c1w1, 8,   64);  packW(i_c1 + 2, c1w2, 64, 32);
    packW(i_c2 + 0, c2w1, 64,  64);  packW(i_c2 + 2, c2w2, 64, 128);
    packW(i_c3 + 0, c3w1, 256, 64);  packW(i_c3 + 2, c3w2, 64, 64);
    packW(i_c4 + 0, c4w1, 128, 64);  packW(i_c4 + 2, c4w2, 64, 128);
    packW(i_in1 + 0, i1w1, 32,  128); packW(i_in1 + 2, i1w2, 128, 32);
    packW(i_in2 + 0, i2w1, 128, 128); packW(i_in2 + 2, i2w2, 128, 128);
    packW(i_in3 + 0, i3w1, 64,  128); packW(i_in3 + 2, i3w2, 128, 64);

    prep_kernel<<<(N + 255) / 256, 256, 0, stream>>>(x, matrix, batch, buf0, N);

    const int egrid = (E / 16 + 7) / 8;   // 8 waves (tiles) per block
    const int ngrid = (N / 16 + 7) / 8;
    auto zero = [&](float* p, long n) {
        zero_kernel<<<(int)((n + 255) / 256), 256, 0, stream>>>(p, n);
    };
    auto fin = [&](float* p, long n) {
        finalize_kernel<<<(int)((n + 255) / 256), 256, 0, stream>>>(p, n);
    };

    // conv1: [N,4] -> [N,32]
    zero(buf1, (long)N * 32);
    conv_kernel<8, 32><<<egrid, 256, 0, stream>>>(buf0, esrc, edst, c1w1, bias(i_c1 + 1),
                                                  c1w2, bias(i_c1 + 3), buf1, E);
    fin(buf1, (long)N * 32);
    // inter1 (32 -> 128 -> 32) + residual
    inter_kernel<32><<<ngrid, 256, 0, stream>>>(buf1, i1w1, bias(i_in1 + 1), i1w2, bias(i_in1 + 3), buf0, N);

    // conv2: [N,32] -> [N,128]
    zero(buf1, (long)N * 128);
    conv_kernel<64, 128><<<egrid, 256, 0, stream>>>(buf0, esrc, edst, c2w1, bias(i_c2 + 1),
                                                    c2w2, bias(i_c2 + 3), buf1, E);
    fin(buf1, (long)N * 128);
    inter_kernel<128><<<ngrid, 256, 0, stream>>>(buf1, i2w1, bias(i_in2 + 1), i2w2, bias(i_in2 + 3), buf0, N);

    // conv3: [N,128] -> [N,64]
    zero(buf1, (long)N * 64);
    conv_kernel<256, 64><<<egrid, 256, 0, stream>>>(buf0, esrc, edst, c3w1, bias(i_c3 + 1),
                                                    c3w2, bias(i_c3 + 3), buf1, E);
    fin(buf1, (long)N * 64);
    inter_kernel<64><<<ngrid, 256, 0, stream>>>(buf1, i3w1, bias(i_in3 + 1), i3w2, bias(i_in3 + 3), buf0, N);

    // conv4: [N,64] -> [N,128]
    zero(buf1, (long)N * 128);
    conv_kernel<128, 128><<<egrid, 256, 0, stream>>>(buf0, esrc, edst, c4w1, bias(i_c4 + 1),
                                                     c4w2, bias(i_c4 + 3), buf1, E);
    fin(buf1, (long)N * 128);

    // pool + fc
    zero(pooled, 256 * 128);
    pool_kernel<<<(int)(((long)N * 128 + 255) / 256), 256, 0, stream>>>(buf1, batch, pooled, N);
    fc_kernel<<<1, 256, 0, stream>>>(pooled, (const float*)d_in[i_fc], (const float*)d_in[i_fc + 1],
                                     (float*)d_out);
}
